// MultiHeadAttention_16484084482771
// MI455X (gfx1250) — compile-verified, matrix operand order checked by reading, not run on hardware
//
#include <hip/hip_runtime.h>

// ---------------------------------------------------------------------------
// MultiHeadAttention (no softmax) for MI455X / gfx1250, wave32, WMMA bf16.
//  * linearity trick: C_h = scale*(V_h K_h^T) Q_h  -> no SxS score matrix
//  * position-major reformulation: every GEMM operand is K-contiguous, so
//    there are ZERO transposes in the inner loops (transposed epilogues and
//    one up-front x transpose handle all layout changes)
//  * tiles staged with GLOBAL_LOAD_ASYNC_TO_LDS_B128 (ASYNCcnt), double
//    buffered so tensor DMA overlaps v_wmma_f32_16x16x32_bf16 issue
// ---------------------------------------------------------------------------

typedef __attribute__((ext_vector_type(16))) __bf16 v16bf;
typedef __attribute__((ext_vector_type(8)))  float  v8f;

// exact pointee type expected by the async-to-LDS builtins (from hipcc diag):
//   __attribute__((__vector_size__(4 * sizeof(int)))) int
typedef int v4i_vs __attribute__((__vector_size__(4 * sizeof(int))));
typedef __attribute__((address_space(1))) v4i_vs glob_v4i;   // prints __device__
typedef __attribute__((address_space(3))) v4i_vs lds_v4i;    // prints __shared__

#define BDIM   4
#define NUNITS 1024
#define SEQ    2048
#define NHEAD  8
#define DHEAD  128
#define SCALE_F 0.08838834764831845f   // 1/sqrt(128)

#define LDSS   40                      // 32 + 8 pad (bf16 elems), 16B aligned
#define AROWS  128                     // block tile M
#define BROWS  64                      // block tile N
#define ATILE  (AROWS * LDSS)          // 5120 elems
#define BTILE  (BROWS * LDSS)          // 2560 elems
#define BUFE   (ATILE + BTILE)         // 7680 elems per buffer

// round-to-nearest-even f32 -> bf16
static __device__ __forceinline__ __bf16 f2bf(float f) {
    unsigned u = __builtin_bit_cast(unsigned, f);
    unsigned r = (u + 0x7FFFu + ((u >> 16) & 1u)) >> 16;
    unsigned short h = (unsigned short)r;
    return __builtin_bit_cast(__bf16, h);
}

// 16-byte global -> LDS copy. CDNA5 path: async DMA straight into LDS
// (no VGPR round trip, tracked by ASYNCcnt). Fallback: sync copy.
static __device__ __forceinline__ void cp16_g2l(const __bf16* g, __bf16* l) {
#if __has_builtin(__builtin_amdgcn_global_load_async_to_lds_b128)
    __builtin_amdgcn_global_load_async_to_lds_b128(
        (glob_v4i*)(g), (lds_v4i*)(l), 0, 0);
#else
    *reinterpret_cast<uint4*>(l) = *reinterpret_cast<const uint4*>(g);
#endif
}

static __device__ __forceinline__ void wait_async0() {
#if __has_builtin(__builtin_amdgcn_s_wait_asynccnt)
    __builtin_amdgcn_s_wait_asynccnt(0);
#else
    asm volatile("s_wait_asynccnt 0x0" ::: "memory");
#endif
}

static __device__ __forceinline__ v8f wmma_bf16(v16bf a, v16bf b, v8f c) {
    return __builtin_amdgcn_wmma_f32_16x16x32_bf16(
        false, a, false, b, (short)0, c, false, false);
}

// Gather one 16x32 bf16 fragment from an LDS tile stored row-major, stride
// LDSS.  ISA layout: lanes 0-15 hold row M=lane, K=0..7 & 16..23; lanes
// 16-31 hold row M=lane-16, K=8..15 & 24..31.  B fragments use the same
// gather (lane <-> column) because all our B operands are NxK row-major.
static __device__ __forceinline__ v16bf load_frag(const __bf16* base, int lane) {
    const int row = lane & 15;
    const int kb  = (lane >> 4) << 3;
    const __bf16* p = base + row * LDSS + kb;
    union { v16bf v; uint4 q[2]; } u;
    u.q[0] = *reinterpret_cast<const uint4*>(p);       // K = kb .. kb+7
    u.q[1] = *reinterpret_cast<const uint4*>(p + 16);  // K = kb+16 .. kb+23
    return u.v;
}

__global__ void cvt_f32_to_bf16(const float* __restrict__ in,
                                __bf16* __restrict__ out, long long n) {
    long long i  = (long long)blockIdx.x * blockDim.x + threadIdx.x;
    long long st = (long long)gridDim.x * blockDim.x;
    for (; i < n; i += st) out[i] = f2bf(in[i]);
}

// x (B, C, S) f32  ->  xT (B, S, C) bf16   (tiled 32x32 via LDS)
__global__ __launch_bounds__(256)
void transpose_cvt(const float* __restrict__ in, __bf16* __restrict__ out) {
    __shared__ float t[32][33];
    const int b  = blockIdx.z;
    const int s0 = blockIdx.x * 32, c0 = blockIdx.y * 32;
    const int tx = threadIdx.x, ty = threadIdx.y;         // (32, 8)
    const float* src = in  + (long long)b * NUNITS * SEQ;
    __bf16*      dst = out + (long long)b * SEQ * NUNITS;
    #pragma unroll
    for (int r = 0; r < 4; ++r)
        t[ty + 8 * r][tx] = src[(long long)(c0 + ty + 8 * r) * SEQ + s0 + tx];
    __syncthreads();
    #pragma unroll
    for (int r = 0; r < 4; ++r)
        dst[(long long)(s0 + ty + 8 * r) * NUNITS + c0 + tx] =
            f2bf(t[tx][ty + 8 * r]);
}

// Batched GEMM  D = alpha * A(MxK) * B^T, bf16 in, f32 accumulate.
//   A: M x K row-major (K contiguous),  B: N x K row-major (K contiguous).
// TROUT=0: store D row-major  (ldc = row stride over M)
// TROUT=1: store D transposed (C[n][m], ldc = row stride over N) using
//          per-lane 16B vector stores (the 8 accumulator rows are contiguous)
// Block tile 128x64, BK=32, 128 threads (4 waves), wave tile 32x64 (2x4 wmma).
// LDS double buffered; next tile staged asynchronously during compute.
template <typename OutT, bool TROUT>
__global__ __launch_bounds__(128)
void gemm_wmma_bf16(const __bf16* __restrict__ A, const __bf16* __restrict__ B,
                    OutT* __restrict__ C,
                    int K, int lda, int ldb, int ldc,
                    long long sAo, long long sAi,
                    long long sBo, long long sBi,
                    long long sCo, long long sCi,
                    int batch_inner, float alpha) {
    __shared__ __align__(16) __bf16 smem[2 * BUFE];

    const int z  = blockIdx.z;
    const int zo = z / batch_inner;
    const int zi = z - zo * batch_inner;
    A += zo * sAo + zi * sAi;
    B += zo * sBo + zi * sBi;
    C += zo * sCo + zi * sCi;

    const int m0 = blockIdx.y * AROWS;
    const int n0 = blockIdx.x * BROWS;
    const int tid  = threadIdx.x;
    const int lane = tid & 31;
    const int wm   = (tid >> 5) * 32;          // wave's 32-row strip of M

    const __bf16* gA = A + (long long)m0 * lda;
    const __bf16* gB = B + (long long)n0 * ldb;

    // staging coordinates: 16B chunks, thread covers rows r0, r0+32, ...
    const int r0  = tid >> 2;                  // 0..31
    const int c8  = (tid & 3) << 3;            // 0,8,16,24

    auto stage = [&](int buf, int k0) {
        __bf16* sA = smem + buf * BUFE;
        __bf16* sB = sA + ATILE;
        #pragma unroll
        for (int rr = 0; rr < AROWS; rr += 32)
            cp16_g2l(&gA[(long long)(r0 + rr) * lda + k0 + c8],
                     &sA[(r0 + rr) * LDSS + c8]);
        #pragma unroll
        for (int rr = 0; rr < BROWS; rr += 32)
            cp16_g2l(&gB[(long long)(r0 + rr) * ldb + k0 + c8],
                     &sB[(r0 + rr) * LDSS + c8]);
    };

    v8f acc[2][4];
    #pragma unroll
    for (int i = 0; i < 2; ++i)
        #pragma unroll
        for (int j = 0; j < 4; ++j) acc[i][j] = (v8f){};

    const int nK = K >> 5;
    stage(0, 0);
    for (int it = 0; it < nK; ++it) {
        wait_async0();          // this wave's async writes have landed in LDS
        __syncthreads();        // everyone's writes visible; prev buf free
        if (it + 1 < nK) stage((it + 1) & 1, (it + 1) << 5);

        const __bf16* sA = smem + (it & 1) * BUFE;
        const __bf16* sB = sA + ATILE;

        v16bf af[2], bf[4];
        #pragma unroll
        for (int i = 0; i < 2; ++i)
            af[i] = load_frag(sA + (wm + 16 * i) * LDSS, lane);
        #pragma unroll
        for (int j = 0; j < 4; ++j)
            bf[j] = load_frag(sB + (16 * j) * LDSS, lane);

        #pragma unroll
        for (int i = 0; i < 2; ++i)
            #pragma unroll
            for (int j = 0; j < 4; ++j)
                acc[i][j] = wmma_bf16(af[i], bf[j], acc[i][j]);
    }

    // epilogue. D layout: VGPR r -> M = r + 8*(lane>=16), N = lane&15
    const int r_off = (lane >> 4) << 3;
    const int cn    = lane & 15;
    #pragma unroll
    for (int i = 0; i < 2; ++i) {
        #pragma unroll
        for (int j = 0; j < 4; ++j) {
            const int gm = m0 + wm + 16 * i + r_off;   // 8 consecutive rows
            const int gn = n0 + 16 * j + cn;
            if (TROUT) {
                // C[gn][gm..gm+7]: contiguous per lane -> 16B vector stores
                const long long base = (long long)gn * ldc + gm;
                if (sizeof(OutT) == 4) {
                    float* Cf = reinterpret_cast<float*>(C);
                    union { uint4 q[2]; float f[8]; } u;
                    #pragma unroll
                    for (int r = 0; r < 8; ++r) u.f[r] = alpha * acc[i][j][r];
                    *reinterpret_cast<uint4*>(&Cf[base])     = u.q[0];
                    *reinterpret_cast<uint4*>(&Cf[base + 4]) = u.q[1];
                } else {
                    __bf16* Cb = reinterpret_cast<__bf16*>(C);
                    union { uint4 q; __bf16 h[8]; } u;
                    #pragma unroll
                    for (int r = 0; r < 8; ++r) u.h[r] = f2bf(alpha * acc[i][j][r]);
                    *reinterpret_cast<uint4*>(&Cb[base]) = u.q;
                }
            } else {
                #pragma unroll
                for (int r = 0; r < 8; ++r) {
                    const float v = alpha * acc[i][j][r];
                    const long long idx = (long long)(gm + r) * ldc + gn;
                    if (sizeof(OutT) == 4)
                        reinterpret_cast<float*>(C)[idx] = v;
                    else
                        reinterpret_cast<__bf16*>(C)[idx] = f2bf(v);
                }
            }
        }
    }
}

extern "C" void kernel_launch(void* const* d_in, const int* in_sizes, int n_in,
                              void* d_out, int out_size, void* d_ws, size_t ws_size,
                              hipStream_t stream) {
    (void)in_sizes; (void)n_in; (void)out_size; (void)ws_size;
    const float* x    = (const float*)d_in[0];   // (4,1024,2048)
    const float* Wqkv = (const float*)d_in[1];   // (3072,1024) row-major [o][c]
    const float* Wout = (const float*)d_in[2];   // (1024,1024) row-major [o][c]
    float*       out  = (float*)d_out;           // (4,1024,2048)

    const long long sX    = (long long)NUNITS * SEQ;          // 2,097,152
    const long long NX    = (long long)BDIM * sX;             // 8,388,608
    const long long NWQKV = (long long)3 * NUNITS * NUNITS;
    const long long NWOUT = (long long)NUNITS * NUNITS;
    const long long sM    = (long long)DHEAD * DHEAD;         // 16384
    const long long sHS   = (long long)DHEAD * SEQ;           // 262144

    char* ws = (char*)d_ws;
    long long o = 0;
    __bf16* xT    = (__bf16*)(ws + o); o += 2 * NX;      // (B, S, C)
    __bf16* wqkvb = (__bf16*)(ws + o); o += 2 * NWQKV;   // (3072, 1024)
    __bf16* woutb = (__bf16*)(ws + o); o += 2 * NWOUT;   // (1024, 1024)
    __bf16* qT    = (__bf16*)(ws + o); o += 2 * NX;      // (B, S, 1024)  S-major
    __bf16* kC    = (__bf16*)(ws + o); o += 2 * NX;      // (B, 1024, S)  C-major
    __bf16* vC    = (__bf16*)(ws + o); o += 2 * NX;      // (B, 1024, S)  C-major
    __bf16* mm    = (__bf16*)(ws + o); o += 2 * (BDIM * NHEAD * sM); // (B,H,d,d')
    __bf16* cT    = (__bf16*)(ws + o);                    // (B, S, 1024) S-major

    // ---- precision casts / layout prep ----
    {
        dim3 tb(32, 8), tg(SEQ / 32, NUNITS / 32, BDIM);
        transpose_cvt<<<tg, tb, 0, stream>>>(x, xT);
    }
    cvt_f32_to_bf16<<<1024, 256, 0, stream>>>(Wqkv, wqkvb, NWQKV);
    cvt_f32_to_bf16<<< 512, 256, 0, stream>>>(Wout, woutb, NWOUT);

    // ---- Pass 1: qkvT = xT @ Wqkv^T, split into Q (S-major) / K,V (C-major)
    dim3 g1(NUNITS / BROWS, SEQ / AROWS, BDIM);              // (16,16,4)
    gemm_wmma_bf16<__bf16, false><<<g1, 128, 0, stream>>>(   // Q
        xT, wqkvb, qT, NUNITS, NUNITS, NUNITS, NUNITS,
        sX, 0, 0, 0, sX, 0, 1, 1.0f);
    gemm_wmma_bf16<__bf16, true><<<g1, 128, 0, stream>>>(    // K -> C-major
        xT, wqkvb + (long long)NUNITS * NUNITS, kC,
        NUNITS, NUNITS, NUNITS, SEQ,
        sX, 0, 0, 0, sX, 0, 1, 1.0f);
    gemm_wmma_bf16<__bf16, true><<<g1, 128, 0, stream>>>(    // V -> C-major
        xT, wqkvb + (long long)2 * NUNITS * NUNITS, vC,
        NUNITS, NUNITS, NUNITS, SEQ,
        sX, 0, 0, 0, sX, 0, 1, 1.0f);

    // ---- Pass 2: M[b,h] = SCALE * V_h @ K_h^T   (both S-contiguous) ----
    dim3 g2(DHEAD / BROWS, DHEAD / AROWS, BDIM * NHEAD);     // (2,1,32)
    gemm_wmma_bf16<__bf16, false><<<g2, 128, 0, stream>>>(
        vC, kC, mm, SEQ, SEQ, SEQ, DHEAD,
        sX, sHS, sX, sHS, (long long)NHEAD * sM, sM, NHEAD, SCALE_F);

    // ---- Pass 3: cT[b,:,h*128+d] = qT_h @ M[b,h]^T ----
    dim3 g3(DHEAD / BROWS, SEQ / AROWS, BDIM * NHEAD);       // (2,16,32)
    gemm_wmma_bf16<__bf16, false><<<g3, 128, 0, stream>>>(
        qT, mm, cT, DHEAD, NUNITS, DHEAD, NUNITS,
        sX, DHEAD, (long long)NHEAD * sM, sM, sX, DHEAD, NHEAD, 1.0f);

    // ---- Pass 4: out[b] = (cT @ Wout^T)^T  -> f32, transposed epilogue ----
    dim3 g4(NUNITS / BROWS, SEQ / AROWS, BDIM);              // (16,16,4)
    gemm_wmma_bf16<float, true><<<g4, 128, 0, stream>>>(
        cT, woutb, out, NUNITS, NUNITS, NUNITS, SEQ,
        sX, 0, 0, 0, sX, 0, 1, 1.0f);
}
